// TransformerConv_56908316672627
// MI455X (gfx1250) — matrix operand order repaired and checked by reading, not compiled, not run on hardware
//
#include <hip/hip_runtime.h>

// ---------------------------------------------------------------------------
// TransformerConv for MI455X (gfx1250, wave32)
//   N=100000 nodes, E=1.6M edges, D_IN=128, HEADS=4, C=32  (HEADS*C = 128)
//
// Pipeline:
//   0. pack weights [Wq|Wk|Wv|Wskip] -> bf16, TRANSPOSED [512][128]  (ws)
//   1. convert x -> bf16 [N][128]                                    (ws)
//   2. init per-node softmax state (ordered-uint max enc, z=0)       (ws)
//   3. fused GEMM [N x 128] x [128 x 512] via v_wmma_f32_16x16x32_bf16
//        -> qb,kb,vb as bf16 (L2-resident gathers later), skip -> d_out (f32)
//      FULL-template: guard-free epilogue for complete row tiles; a separate
//      single-block guarded launch handles any N%16 tail (host-side split).
//   4. edge logits: wave/edge (scalar edge idx), lane = 4 channels,
//        8-lane butterfly per head, atomicMax on ordered-u32 for segment max
//   5. edge exp:    thread/edge, vectorized p=exp(a-m), atomicAdd z
//   6. edge scatter: wave/edge, b64 v-gather, 4 f32 atomics/lane into out[dst]
// ---------------------------------------------------------------------------

typedef __attribute__((ext_vector_type(16))) __bf16       v16bf;
typedef __attribute__((ext_vector_type(8)))  float        v8f;
typedef __attribute__((ext_vector_type(4)))  float        f32x4;
typedef __attribute__((ext_vector_type(4)))  unsigned int u32x4;
typedef __attribute__((ext_vector_type(2)))  unsigned int u32x2;

union Frag { v16bf v; u32x4 u[2]; };

static constexpr int D  = 128;   // input dim == HEADS*C
static constexpr int H  = 4;
static constexpr unsigned ENC_NEG_INF = 0x007FFFFFu; // enc(-inf)

__device__ __forceinline__ unsigned short f2bf(float f) {
  unsigned b = __builtin_bit_cast(unsigned, f);
  b += 0x7FFFu + ((b >> 16) & 1u);            // round-to-nearest-even
  return (unsigned short)(b >> 16);
}
__device__ __forceinline__ float bf_lo(unsigned u) {   // low bf16 of a dword
  return __builtin_bit_cast(float, u << 16);
}
__device__ __forceinline__ float bf_hi(unsigned u) {   // high bf16 of a dword
  return __builtin_bit_cast(float, u & 0xFFFF0000u);
}
// monotone float <-> uint mapping for atomic max on signed floats
__device__ __forceinline__ unsigned encf(float f) {
  unsigned b = __builtin_bit_cast(unsigned, f);
  return (b & 0x80000000u) ? ~b : (b | 0x80000000u);
}
__device__ __forceinline__ float decf(unsigned e) {
  unsigned b = (e & 0x80000000u) ? (e ^ 0x80000000u) : ~e;
  return __builtin_bit_cast(float, b);
}

// ---- 0: pack weights (bf16, transposed [512][128]) + biases -----------------
__global__ __launch_bounds__(256) void pack_weights(
    const float* __restrict__ Wq, const float* __restrict__ bq,
    const float* __restrict__ Wk, const float* __restrict__ bk,
    const float* __restrict__ Wv, const float* __restrict__ bv,
    const float* __restrict__ Ws, const float* __restrict__ bs,
    unsigned short* __restrict__ wT, float* __restrict__ bcat) {
  int tid = blockIdx.x * 256 + threadIdx.x;       // over 512*128
  if (tid >= 512 * D) return;
  int j = tid / D;            // global output col 0..511
  int k = tid % D;            // input dim
  int m = j >> 7;             // which matrix
  int lc = j & 127;           // col within matrix
  const float* W = (m == 0) ? Wq : (m == 1) ? Wk : (m == 2) ? Wv : Ws;
  const float* B = (m == 0) ? bq : (m == 1) ? bk : (m == 2) ? bv : bs;
  wT[(size_t)j * D + k] = f2bf(W[(size_t)k * D + lc]);   // transpose
  if (k == 0) bcat[j] = B[lc];
}

// ---- 1: x -> bf16 -----------------------------------------------------------
__global__ __launch_bounds__(256) void convert_x(
    const float* __restrict__ x, unsigned short* __restrict__ xb, long long n) {
  long long t = (long long)blockIdx.x * 256 + threadIdx.x;
  if (t < n) xb[t] = f2bf(x[t]);
}

// ---- 2: init per-node softmax state ----------------------------------------
__global__ __launch_bounds__(256) void init_nodes(
    unsigned* __restrict__ menc, float* __restrict__ z, int n4) {
  int t = blockIdx.x * 256 + threadIdx.x;
  if (t < n4) { menc[t] = ENC_NEG_INF; z[t] = 0.0f; }
}

// ---- 3: fused q/k/v/skip GEMM with bf16 WMMA --------------------------------
// block = 128 threads (4 waves); block computes rows [row0, row0+16) x 512 cols.
// Each wave owns 8 column tiles of 16; K-loop = 128/32 = 4 WMMA per tile.
// FULL=true: all 16 rows valid, no guards. FULL=false: tail block (row guard).
template <bool FULL>
__global__ __launch_bounds__(128) void gemm_qkvs(
    const unsigned short* __restrict__ xb,   // [N][128] bf16
    const unsigned short* __restrict__ wT,   // [512][128] bf16 (transposed)
    const float* __restrict__ bcat,          // [512]
    unsigned short* __restrict__ qb, unsigned short* __restrict__ kb,
    unsigned short* __restrict__ vb, float* __restrict__ out,
    int rowBase, int N) {
  const int wave = threadIdx.x >> 5;
  const int lane = threadIdx.x & 31;
  const int half = lane >> 4;        // 0: lanes 0-15, 1: lanes 16-31
  const int l16  = lane & 15;
  const int row0 = rowBase + blockIdx.x * 16;  // uniform

  // A fragments for the 4 K-tiles (row = row0 + l16; ISA 16-bit A layout:
  // elems 0-7 -> K 0..7 (+8*half), elems 8-15 -> K 16..23 (+8*half))
  int arow = row0 + l16;
  if (!FULL && arow >= N) arow = N - 1;
  const unsigned short* xrow = xb + (size_t)arow * D;
  Frag afrag[4];
#pragma unroll
  for (int kt = 0; kt < 4; ++kt) {
    const int k0 = kt * 32;
    afrag[kt].u[0] = *(const u32x4*)(xrow + k0 + 8 * half);
    afrag[kt].u[1] = *(const u32x4*)(xrow + k0 + 16 + 8 * half);
  }

#pragma unroll
  for (int t = 0; t < 8; ++t) {
    const int n0 = (wave * 8 + t) * 16;           // global col tile base
    const unsigned short* wcol = wT + (size_t)(n0 + l16) * D + 8 * half;
    v8f acc = {};
#pragma unroll
    for (int kt = 0; kt < 4; ++kt) {
      Frag bfrag;                                  // B: lane=col, same K layout as A (0-7/16-23 +8*half)
      bfrag.u[0] = *(const u32x4*)(wcol + kt * 32);
      bfrag.u[1] = *(const u32x4*)(wcol + kt * 32 + 16);
      acc = __builtin_amdgcn_wmma_f32_16x16x32_bf16(
          false, afrag[kt].v, false, bfrag.v, (short)0, acc, false, false);
    }
    const int gcol = n0 + l16;
    const float bias = bcat[gcol];
    // n0 is wave-uniform; force the matrix select onto the scalar path
    const int mtx = __builtin_amdgcn_readfirstlane(n0) >> 7; // 0=q 1=k 2=v 3=skip
    const int lcol = gcol & 127;
    const int nodeBase = row0 + 8 * half;          // C layout: M = r + 8*half
    if (mtx == 3) {                // skip connection -> f32 into d_out
      float* op = out + (size_t)nodeBase * D + lcol;
#pragma unroll
      for (int r = 0; r < 8; ++r) {
        if (FULL || nodeBase + r < N) op[(size_t)r * D] = acc[r] + bias;
      }
    } else {                       // q/k/v -> bf16 scratch (L2-resident)
      unsigned short* dp0 = (mtx == 0) ? qb : (mtx == 1) ? kb : vb;
      unsigned short* dp = dp0 + (size_t)nodeBase * D + lcol;
#pragma unroll
      for (int r = 0; r < 8; ++r) {
        if (FULL || nodeBase + r < N) dp[(size_t)r * D] = f2bf(acc[r] + bias);
      }
    }
  }
}

// ---- 4: per-edge attention logits + segment max -----------------------------
// Wave per edge; edge index forced scalar -> SMEM loads + saddr-form gathers.
// Lane owns channels [4*lane, 4*lane+4); head = lane>>3.
__global__ __launch_bounds__(256) void edge_logits(
    const int* __restrict__ ei, const unsigned short* __restrict__ qb,
    const unsigned short* __restrict__ kb, float* __restrict__ p,
    unsigned* __restrict__ menc, int E) {
  const int e = __builtin_amdgcn_readfirstlane(blockIdx.x * 8 + (threadIdx.x >> 5));
  if (e >= E) return;
  const int lane = threadIdx.x & 31;
  const int src = ei[e];          // scalar load
  const int dst = ei[E + e];      // scalar load
  const u32x2 qv = *(const u32x2*)(qb + (size_t)dst * D + 4 * lane);
  const u32x2 kv = *(const u32x2*)(kb + (size_t)src * D + 4 * lane);
  float s = bf_lo(qv[0]) * bf_lo(kv[0]) + bf_hi(qv[0]) * bf_hi(kv[0]) +
            bf_lo(qv[1]) * bf_lo(kv[1]) + bf_hi(qv[1]) * bf_hi(kv[1]);
  // reduce within each 8-lane group (== one head, all 4 heads in parallel)
  s += __shfl_xor(s, 1, 32);
  s += __shfl_xor(s, 2, 32);
  s += __shfl_xor(s, 4, 32);
  s *= 0.17677669529663687f;      // 1/sqrt(32)
  const int h = lane >> 3;
  if ((lane & 7) == 0) {
    p[(size_t)e * H + h] = s;
    atomicMax(&menc[(size_t)dst * H + h], encf(s));
  }
}

// ---- 5: p = exp(alpha - m); z += p  (thread per edge, vectorized) -----------
__global__ __launch_bounds__(256) void edge_exp(
    const int* __restrict__ ei, float* __restrict__ p,
    const unsigned* __restrict__ menc, float* __restrict__ z, int E) {
  const int e = blockIdx.x * 256 + threadIdx.x;
  if (e >= E) return;
  const int dst = ei[E + e];
  const f32x4 a = *(const f32x4*)(p + (size_t)e * H);
  const u32x4 me = *(const u32x4*)(menc + (size_t)dst * H);
  f32x4 r;
  r[0] = __expf(a[0] - decf(me[0]));
  r[1] = __expf(a[1] - decf(me[1]));
  r[2] = __expf(a[2] - decf(me[2]));
  r[3] = __expf(a[3] - decf(me[3]));
  *(f32x4*)(p + (size_t)e * H) = r;
  float* zp = z + (size_t)dst * H;
  atomicAdd(zp + 0, r[0]);
  atomicAdd(zp + 1, r[1]);
  atomicAdd(zp + 2, r[2]);
  atomicAdd(zp + 3, r[3]);
}

// ---- 6: weighted scatter of v -----------------------------------------------
// Wave per edge (scalar edge idx); lane owns 4 channels; L2-resident atomics.
__global__ __launch_bounds__(256) void edge_scatter(
    const int* __restrict__ ei, const float* __restrict__ p,
    const float* __restrict__ z, const unsigned short* __restrict__ vb,
    float* __restrict__ out, int E) {
  const int e = __builtin_amdgcn_readfirstlane(blockIdx.x * 8 + (threadIdx.x >> 5));
  if (e >= E) return;
  const int lane = threadIdx.x & 31;
  const int src = ei[e];          // scalar load
  const int dst = ei[E + e];      // scalar load
  const int h = lane >> 3;
  const float w = p[(size_t)e * H + h] / (z[(size_t)dst * H + h] + 1e-16f);
  const u32x2 vv = *(const u32x2*)(vb + (size_t)src * D + 4 * lane);
  float* op = out + (size_t)dst * D + 4 * lane;
  atomicAdd(op + 0, w * bf_lo(vv[0]));
  atomicAdd(op + 1, w * bf_hi(vv[0]));
  atomicAdd(op + 2, w * bf_lo(vv[1]));
  atomicAdd(op + 3, w * bf_hi(vv[1]));
}

// ---------------------------------------------------------------------------
extern "C" void kernel_launch(void* const* d_in, const int* in_sizes, int n_in,
                              void* d_out, int out_size, void* d_ws, size_t ws_size,
                              hipStream_t stream) {
  const float* x  = (const float*)d_in[0];
  const int*   ei = (const int*)d_in[1];
  const float* Wq = (const float*)d_in[2];
  const float* bq = (const float*)d_in[3];
  const float* Wk = (const float*)d_in[4];
  const float* bk = (const float*)d_in[5];
  const float* Wv = (const float*)d_in[6];
  const float* bv = (const float*)d_in[7];
  const float* Ws = (const float*)d_in[8];
  const float* bs = (const float*)d_in[9];
  float* out = (float*)d_out;

  const int N = in_sizes[0] / D;
  const int E = in_sizes[1] / 2;

  // workspace carve-up (256B aligned)
  char* ws = (char*)d_ws;
  size_t off = 0;
  auto carve = [&](size_t bytes) -> void* {
    void* ptr = ws + off;
    off = (off + bytes + 255) & ~(size_t)255;
    return ptr;
  };
  unsigned short* xb   = (unsigned short*)carve((size_t)N * D * 2);
  unsigned short* qb   = (unsigned short*)carve((size_t)N * D * 2);
  unsigned short* kb   = (unsigned short*)carve((size_t)N * D * 2);
  unsigned short* vb   = (unsigned short*)carve((size_t)N * D * 2);
  unsigned short* wT   = (unsigned short*)carve((size_t)512 * D * 2);
  float*          bcat = (float*)carve(512 * sizeof(float));
  unsigned*       menc = (unsigned*)carve((size_t)N * H * sizeof(unsigned));
  float*          zbuf = (float*)carve((size_t)N * H * sizeof(float));
  float*          pbuf = (float*)carve((size_t)E * H * sizeof(float));

  // 0: weights -> bf16 transposed
  pack_weights<<<(512 * D + 255) / 256, 256, 0, stream>>>(Wq, bq, Wk, bk, Wv, bv,
                                                          Ws, bs, wT, bcat);
  // 1: x -> bf16
  const long long nx = (long long)N * D;
  convert_x<<<(unsigned)((nx + 255) / 256), 256, 0, stream>>>(x, xb, nx);
  // 2: init segment-softmax state
  init_nodes<<<(N * H + 255) / 256, 256, 0, stream>>>(menc, zbuf, N * H);
  // 3: fused qkv+skip GEMM (WMMA bf16); writes skip into d_out.
  //    Host-side split: guard-free kernel for complete 16-row tiles,
  //    one guarded block for the tail (N=100000 -> no tail).
  const int nFull = N / 16;
  if (nFull > 0)
    gemm_qkvs<true><<<nFull, 128, 0, stream>>>(xb, wT, bcat, qb, kb, vb, out, 0, N);
  if (N % 16)
    gemm_qkvs<false><<<1, 128, 0, stream>>>(xb, wT, bcat, qb, kb, vb, out,
                                            nFull * 16, N);
  // 4-6: edge passes
  edge_logits<<<(E + 7) / 8, 256, 0, stream>>>(ei, qb, kb, pbuf, menc, E);
  edge_exp<<<(E + 255) / 256, 256, 0, stream>>>(ei, pbuf, menc, zbuf, E);
  edge_scatter<<<(E + 7) / 8, 256, 0, stream>>>(ei, pbuf, zbuf, vb, out, E);
}
